// SemanticLayer_24077586661958
// MI455X (gfx1250) — compile-verified, matrix operand order checked by reading, not run on hardware
//
#include <hip/hip_runtime.h>
#include <hip/hip_bf16.h>
#include <math.h>

// ---------------------------------------------------------------------------
// MI455X (gfx1250) implementation.
//  * All GEMMs (GRU gates + relation-basis conv) run on v_wmma_f32_16x16x32_bf16.
//  * Weights strip staged into LDS via async global->LDS copies (ASYNCcnt),
//    register-blocked 16x64 per wave (4 accumulators, A-frag reuse 4x).
//  * B fragments for a k-step are preloaded as a clause before the 4 WMMAs so
//    only one s_wait_dscnt precedes the back-to-back matrix ops.
//  * RelGraphConv: per relation r, tmp = h @ W_r (dense WMMA GEMM), then an
//    atomic scatter over the edge list adds norm-weighted rows into the output.
// ---------------------------------------------------------------------------

typedef __bf16 bf16_t;
typedef __attribute__((ext_vector_type(8)))  bf16_t bf16x8;
typedef __attribute__((ext_vector_type(16))) bf16_t bf16x16;
typedef __attribute__((ext_vector_type(8)))  float  f32x8;

#define NENT   60000
#define NB     20000     // NENT / BATCH
#define HD     128
#define GDIM   384       // 3 * OUT_DIM
#define NREL   11
#define NBASES 22
#define NEDGE  300000
#define LDSK   136       // padded K stride (elements): 272B row -> conflict-free banks

// ---- f32 -> bf16 conversion (for WMMA operands) ---------------------------
__global__ void cvt_f32_bf16(const float* __restrict__ src,
                             bf16_t* __restrict__ dst, int n) {
  int i = blockIdx.x * blockDim.x + threadIdx.x;
  if (i < n) dst[i] = (bf16_t)src[i];
}

// ---- WMMA GEMM: C[M,N] = A[M,K] * Bw[N,K]^T, bf16 in / f32 out ------------
// Block: 128 threads = 4 waves. Block tile: 64(M) x 64(N), K=128.
// B strip (64 rows x K) staged to LDS once via global_load_async_to_lds_b128.
// Each wave owns a 16x64 C strip: 4 f32x8 accumulators, A fragment reused 4x.
// A fragment layout per CDNA5 ISA 7.12.2 (16-bit A 16x32):
//   lanes 0-15  : row m = lane,    K in {0..7} u {16..23}
//   lanes 16-31 : row m = lane-16, K in {8..15} u {24..31}
__global__ __launch_bounds__(128)
void wmma_gemm_bf16(const bf16_t* __restrict__ A, const bf16_t* __restrict__ Bw,
                    float* __restrict__ C, int M, int N, int K) {
  __shared__ bf16_t sB[64 * LDSK];               // 64 N-rows x padded K
  const int tid  = threadIdx.x;
  const int wave = tid >> 5;
  const int lane = tid & 31;
  const int n0   = blockIdx.y << 6;

  // ---- async-stage B strip rows [n0, n0+64) x K=128 into LDS --------------
  {
    const bf16_t* gB = Bw + (size_t)n0 * K;      // 64 rows x 256B, contiguous
    unsigned lbase = (unsigned)(size_t)(&sB[0]); // low 32 bits = LDS offset
    #pragma unroll
    for (int i = 0; i < 8; ++i) {                // 1024 16B chunks / 128 thr
      int c   = i * 128 + tid;
      int row = c >> 4;                          // 0..63
      int col = c & 15;                          // 0..15 (16B chunks)
      unsigned           laddr = lbase + (unsigned)(row * (LDSK * 2) + col * 16);
      unsigned long long gaddr = (unsigned long long)gB + (unsigned)(row * 256 + col * 16);
      asm volatile("global_load_async_to_lds_b128 %0, %1, off"
                   :: "v"(laddr), "v"(gaddr) : "memory");
    }
    asm volatile("s_wait_asynccnt 0x0" ::: "memory");
  }
  __syncthreads();

  const int Mtiles = M >> 4;
  const int mtile  = blockIdx.x * 4 + wave;      // wave-uniform
  if (mtile >= Mtiles) return;                   // whole wave exits together
  const int m0   = mtile << 4;
  const int half = lane >> 4;                    // 0 or 1
  const int l15  = lane & 15;

  const bf16_t* ap = A + (size_t)(m0 + l15) * K + (half << 3);
  __builtin_prefetch(ap, 0, 1);                  // gfx1250 global_prefetch_b8

  f32x8 acc[4];
  const f32x8 zero = {0.f, 0.f, 0.f, 0.f, 0.f, 0.f, 0.f, 0.f};
  acc[0] = zero; acc[1] = zero; acc[2] = zero; acc[3] = zero;

  const bf16_t* lb = &sB[(size_t)l15 * LDSK + (half << 3)];
  #pragma unroll 4
  for (int k0 = 0; k0 < K; k0 += 32) {
    // -- issue all loads for this k-step first (one clause, one wait) --
    bf16x8 al = *(const bf16x8*)(ap + k0);
    bf16x8 ah = *(const bf16x8*)(ap + k0 + 16);
    bf16x8 bl[4], bh[4];
    #pragma unroll
    for (int nf = 0; nf < 4; ++nf) {
      const bf16_t* bp = lb + nf * 16 * LDSK + k0;
      bl[nf] = *(const bf16x8*)(bp);
      bh[nf] = *(const bf16x8*)(bp + 16);
    }
    bf16x16 a = __builtin_shufflevector(al, ah, 0,1,2,3,4,5,6,7,8,9,10,11,12,13,14,15);
    // -- 4 independent WMMAs back-to-back (same A, disjoint accumulators) --
    #pragma unroll
    for (int nf = 0; nf < 4; ++nf) {
      bf16x16 b = __builtin_shufflevector(bl[nf], bh[nf],
                                          0,1,2,3,4,5,6,7,8,9,10,11,12,13,14,15);
      acc[nf] = __builtin_amdgcn_wmma_f32_16x16x32_bf16(
          /*neg_a=*/false, a, /*neg_b=*/false, b,
          /*c_mod=*/(short)0, acc[nf], /*reuse_a=*/false, /*reuse_b=*/false);
    }
  }

  // C/D layout: VGPR r -> row m0 + r + 8*half, lane&15 -> column
  float* cbase = C + (size_t)(m0 + (half << 3)) * N + n0 + l15;
  #pragma unroll
  for (int nf = 0; nf < 4; ++nf)
    #pragma unroll
    for (int r = 0; r < 8; ++r)
      cbase[(size_t)r * N + nf * 16] = acc[nf][r];
}

// ---- GRU gate combine: h' = (1-z)*n + z*h ---------------------------------
__global__ void gru_combine(const float* __restrict__ Gi, const float* __restrict__ bih,
                            const float* __restrict__ Gh, const float* __restrict__ bhh,
                            float* __restrict__ hstate, float* __restrict__ yout, int nb) {
  int idx = blockIdx.x * blockDim.x + threadIdx.x;
  if (idx >= nb * HD) return;
  int b = idx >> 7, j = idx & 127;
  size_t gb = (size_t)b * GDIM;
  float ir = Gi[gb + j]       + bih[j];
  float iz = Gi[gb + 128 + j] + bih[128 + j];
  float in = Gi[gb + 256 + j] + bih[256 + j];
  float hr = Gh[gb + j]       + bhh[j];
  float hz = Gh[gb + 128 + j] + bhh[128 + j];
  float hn = Gh[gb + 256 + j] + bhh[256 + j];
  float r = 1.f / (1.f + __expf(-(ir + hr)));
  float z = 1.f / (1.f + __expf(-(iz + hz)));
  float n = tanhf(in + r * hn);
  float hnext = (1.f - z) * n + z * hstate[idx];
  hstate[idx] = hnext;
  if (yout) yout[idx] = hnext;
}

// ---- basis decomposition: Wt[r][o][i] = sum_b comp[r,b] * V[b,i,o] (bf16) -
// Thread id maps o contiguous -> the 22 V reads per thread are coalesced.
__global__ void basis_weights(const float* __restrict__ comp, const float* __restrict__ V,
                              bf16_t* __restrict__ Wt) {
  int idx = blockIdx.x * blockDim.x + threadIdx.x;  // r*128*128 + i*128 + o
  if (idx >= NREL * HD * HD) return;
  int r = idx >> 14;
  int i = (idx >> 7) & 127;
  int o = idx & 127;
  float s = 0.f;
  #pragma unroll
  for (int b = 0; b < NBASES; ++b)
    s += comp[r * NBASES + b] * V[((size_t)b * HD + i) * HD + o];
  Wt[((size_t)r * HD + o) * HD + i] = (bf16_t)s;   // store transposed [r][o][i]
}

// ---- in-degree + inverse norm ---------------------------------------------
__global__ void degree_kernel(const int* __restrict__ dst, float* __restrict__ deg, int E) {
  int e = blockIdx.x * blockDim.x + threadIdx.x;
  if (e < E) atomicAdd(&deg[dst[e]], 1.0f);
}
__global__ void inv_kernel(const float* __restrict__ deg, float* __restrict__ inv, int n) {
  int i = blockIdx.x * blockDim.x + threadIdx.x;
  if (i < n) { float d = deg[i]; inv[i] = d > 0.f ? 1.f / d : 0.f; }
}

// ---- edge scatter for one relation: out[dst] += tmp[src] * inv[dst] -------
__global__ __launch_bounds__(256)
void scatter_rel(const float* __restrict__ tmp, const int* __restrict__ src,
                 const int* __restrict__ dst, const int* __restrict__ rel,
                 const float* __restrict__ inv, float* __restrict__ out,
                 int r, int E) {
  int gid  = blockIdx.x * blockDim.x + threadIdx.x;
  int e    = gid >> 5;
  int lane = gid & 31;
  if (e >= E || rel[e] != r) return;
  int s = src[e], d = dst[e];
  float w = inv[d];
  const float* tp = tmp + (size_t)s * HD;
  float* op = out + (size_t)d * HD;
  #pragma unroll
  for (int j = 0; j < 4; ++j) {
    int c = lane + j * 32;
    atomicAdd(&op[c], tp[c] * w);
  }
}

// ---- bias add / bias+relu -------------------------------------------------
__global__ void add_bias(float* __restrict__ h, const float* __restrict__ bias, int rows) {
  int idx = blockIdx.x * blockDim.x + threadIdx.x;
  if (idx < rows * HD) h[idx] += bias[idx & 127];
}
__global__ void bias_relu(float* __restrict__ h, const float* __restrict__ bias, int rows) {
  int idx = blockIdx.x * blockDim.x + threadIdx.x;
  if (idx < rows * HD) {
    float v = h[idx] + bias[idx & 127];
    h[idx] = v > 0.f ? v : 0.f;
  }
}

// ---------------------------------------------------------------------------
extern "C" void kernel_launch(void* const* d_in, const int* in_sizes, int n_in,
                              void* d_out, int out_size, void* d_ws, size_t ws_size,
                              hipStream_t stream) {
  (void)in_sizes; (void)n_in; (void)out_size; (void)ws_size;
  const float* ent_emb = (const float*)d_in[0];
  const float* h0      = (const float*)d_in[2];
  const float* Wih[2]  = {(const float*)d_in[3], (const float*)d_in[7]};
  const float* Whh[2]  = {(const float*)d_in[4], (const float*)d_in[8]};
  const float* bih[2]  = {(const float*)d_in[5], (const float*)d_in[9]};
  const float* bhh[2]  = {(const float*)d_in[6], (const float*)d_in[10]};
  const float* comp1 = (const float*)d_in[11];
  const float* V1    = (const float*)d_in[12];
  const float* bias1 = (const float*)d_in[13];
  const float* comp2 = (const float*)d_in[14];
  const float* V2    = (const float*)d_in[15];
  const float* bias2 = (const float*)d_in[16];
  const int* src = (const int*)d_in[17];
  const int* dst = (const int*)d_in[18];
  const int* rel = (const int*)d_in[19];
  float* out = (float*)d_out;

  // ---- workspace carve-out ----
  char* w = (char*)d_ws;
  auto carve = [&](size_t bytes) { char* p = w; w += (bytes + 255) & ~(size_t)255; return p; };
  float*  GI    = (float*)carve((size_t)NB * GDIM * 4);
  float*  GH    = (float*)carve((size_t)NB * GDIM * 4);
  float*  HL0   = (float*)carve((size_t)NB * HD * 4);
  float*  HL1   = (float*)carve((size_t)NB * HD * 4);
  bf16_t* XB    = (bf16_t*)carve((size_t)NB * HD * 2);
  bf16_t* HB    = (bf16_t*)carve((size_t)NB * HD * 2);
  float*  HFULL = (float*)carve((size_t)NENT * HD * 4);
  bf16_t* HFB   = (bf16_t*)carve((size_t)NENT * HD * 2);
  bf16_t* Wihb[2] = {(bf16_t*)carve(GDIM * HD * 2), (bf16_t*)carve(GDIM * HD * 2)};
  bf16_t* Whhb[2] = {(bf16_t*)carve(GDIM * HD * 2), (bf16_t*)carve(GDIM * HD * 2)};
  bf16_t* WT1   = (bf16_t*)carve((size_t)NREL * HD * HD * 2);
  bf16_t* WT2   = (bf16_t*)carve((size_t)NREL * HD * HD * 2);
  float*  DEG   = (float*)carve((size_t)NENT * 4);
  float*  INV   = (float*)carve((size_t)NENT * 4);
  float*  TMP   = (float*)carve((size_t)NENT * HD * 4);
  float*  CONV  = (float*)carve((size_t)NENT * HD * 4);

  auto cvt = [&](const float* s, bf16_t* d, int n) {
    cvt_f32_bf16<<<(n + 255) / 256, 256, 0, stream>>>(s, d, n);
  };
  auto gemm = [&](const bf16_t* A, const bf16_t* B, float* C, int M, int N) {
    dim3 grid(((M >> 4) + 3) / 4, N >> 6);   // waves of 16 M-rows; 64-wide N strips
    wmma_gemm_bf16<<<grid, 128, 0, stream>>>(A, B, C, M, N, HD);
  };

  // ---- weights to bf16 ----
  for (int l = 0; l < 2; ++l) {
    cvt(Wih[l], Wihb[l], GDIM * HD);
    cvt(Whh[l], Whhb[l], GDIM * HD);
  }

  // ---- initial GRU states ----
  hipMemcpyAsync(HL0, h0,                   (size_t)NB * HD * 4, hipMemcpyDeviceToDevice, stream);
  hipMemcpyAsync(HL1, h0 + (size_t)NB * HD, (size_t)NB * HD * 4, hipMemcpyDeviceToDevice, stream);

  // ---- 2-layer GRU, T=3 steps (sequential; batch of 20000 rows parallel) ----
  const int combBlocks = (NB * HD + 255) / 256;
  for (int t = 0; t < 3; ++t) {
    // layer 0
    cvt(ent_emb + (size_t)t * NB * HD, XB, NB * HD);
    gemm(XB, Wihb[0], GI, NB, GDIM);
    cvt(HL0, HB, NB * HD);
    gemm(HB, Whhb[0], GH, NB, GDIM);
    gru_combine<<<combBlocks, 256, 0, stream>>>(GI, bih[0], GH, bhh[0], HL0, nullptr, NB);
    // layer 1 (consumes y0[t] == HL0)
    cvt(HL0, XB, NB * HD);
    gemm(XB, Wihb[1], GI, NB, GDIM);
    cvt(HL1, HB, NB * HD);
    gemm(HB, Whhb[1], GH, NB, GDIM);
    gru_combine<<<combBlocks, 256, 0, stream>>>(GI, bih[1], GH, bhh[1], HL1,
                                                HFULL + (size_t)t * NB * HD, NB);
  }

  // ---- edge norm: 1 / in-degree(dst) ----
  hipMemsetAsync(DEG, 0, (size_t)NENT * 4, stream);
  degree_kernel<<<(NEDGE + 255) / 256, 256, 0, stream>>>(dst, DEG, NEDGE);
  inv_kernel<<<(NENT + 255) / 256, 256, 0, stream>>>(DEG, INV, NENT);

  const int wB = (NREL * HD * HD + 255) / 256;
  const int scatB = (NEDGE * 32 + 255) / 256;

  // ---- RelGraphConv layer 1 ----
  basis_weights<<<wB, 256, 0, stream>>>(comp1, V1, WT1);
  cvt(HFULL, HFB, NENT * HD);
  hipMemsetAsync(CONV, 0, (size_t)NENT * HD * 4, stream);
  for (int r = 0; r < NREL; ++r) {
    gemm(HFB, WT1 + (size_t)r * HD * HD, TMP, NENT, HD);
    scatter_rel<<<scatB, 256, 0, stream>>>(TMP, src, dst, rel, INV, CONV, r, NEDGE);
  }
  add_bias<<<(NENT * HD + 255) / 256, 256, 0, stream>>>(CONV, bias1, NENT);

  // ---- RelGraphConv layer 2 + relu ----
  basis_weights<<<wB, 256, 0, stream>>>(comp2, V2, WT2);
  cvt(CONV, HFB, NENT * HD);
  hipMemsetAsync(out, 0, (size_t)NENT * HD * 4, stream);
  for (int r = 0; r < NREL; ++r) {
    gemm(HFB, WT2 + (size_t)r * HD * HD, TMP, NENT, HD);
    scatter_rel<<<scatB, 256, 0, stream>>>(TMP, src, dst, rel, INV, out, r, NEDGE);
  }
  bias_relu<<<(NENT * HD + 255) / 256, 256, 0, stream>>>(out, bias2, NENT);
}
